// PathFusionEmbedding_51934744543603
// MI455X (gfx1250) — matrix operand order repaired and checked by reading, not compiled, not run on hardware
//
#include <hip/hip_runtime.h>
#include <hip/hip_bf16.h>
#include <math.h>

typedef __attribute__((ext_vector_type(16))) __bf16 v16bf;
typedef __attribute__((ext_vector_type(8)))  float  v8f;

#define BT    51200     // B*T
#define TT    100
#define LL    256
#define DD    128
#define PP    9
#define XPAD  136       // padded LDS row stride (bf16 elems): 272B -> conflict-free

// workspace layout (bytes)
#define WS_WIH   0                         // 512*128 bf16 (gate-permuted, k-contig)
#define WS_WHH   (WS_WIH + 512*128*2)
#define WS_BIAS  (WS_WHH + 512*128*2)      // 512 f32 (permuted b_ih+b_hh)
#define WS_NIDS  (WS_BIAS + 512*4)         // BT*9 int
#define WS_ACT   (WS_NIDS + BT*PP*4)       // BT int

// branch-free tanh: (e-1)/(e+1), e = exp(2*clamp(x,+-15)); exact saturation in fp32
__device__ __forceinline__ float fast_tanh(float x) {
    float xc = fminf(fmaxf(x, -15.0f), 15.0f);
    float e  = __expf(2.0f * xc);
    return (e - 1.0f) * __frcp_rn(e + 1.0f);
}
__device__ __forceinline__ float fast_sigmoid(float x) {
    return __frcp_rn(1.0f + __expf(-x));
}

// ---------------- kernel 0: weight permute + bf16 convert ----------------
__global__ __launch_bounds__(256) void prep_kernel(
    const float* __restrict__ wih, const float* __restrict__ whh,
    const float* __restrict__ bih, const float* __restrict__ bhh,
    __bf16* __restrict__ obih, __bf16* __restrict__ obhh, float* __restrict__ obias)
{
    int idx = blockIdx.x * 256 + threadIdx.x;     // 512 blocks -> 131072 threads
    if (idx < 2 * 512 * 128) {
        int m  = idx >> 16;                       // 0 = ih, 1 = hh
        int r  = idx & 0xFFFF;
        int cp = r >> 7;                          // permuted column 0..511
        int k  = r & 127;
        int n  = (cp & 3) * 128 + (cp >> 2);      // source gate-major row
        float v = (m ? whh : wih)[n * 128 + k];
        (m ? obhh : obih)[cp * 128 + k] = (__bf16)v;
    }
    if (idx < 512) {
        int n = (idx & 3) * 128 + (idx >> 2);
        obias[idx] = bih[n] + bhh[n];
    }
}

// ---------------- kernel 1: argmax leaf + path gather ----------------
__global__ __launch_bounds__(128) void leafpath_kernel(
    const float* __restrict__ cross, const int* __restrict__ paths,
    int* __restrict__ nids, int* __restrict__ act)
{
    int s    = blockIdx.x * 4 + (threadIdx.x >> 5);   // one wave per (b,t)
    int lane = threadIdx.x & 31;
    const float* src = cross + (size_t)s * LL;
    float bv = -3.0e38f; int bi = 0;
    #pragma unroll
    for (int j = 0; j < 8; ++j) {
        int l = lane + 32 * j;                        // ascending l: strict > keeps first max
        float v = src[l];
        if (v > bv) { bv = v; bi = l; }
    }
    #pragma unroll
    for (int off = 16; off >= 1; off >>= 1) {         // lex reduce: max val, then min index
        float ov = __shfl_xor(bv, off, 32);
        int   oi = __shfl_xor(bi, off, 32);
        if (ov > bv || (ov == bv && oi < bi)) { bv = ov; bi = oi; }
    }
    int t = s % TT;
    if (lane < PP)
        nids[s * PP + lane] = paths[((size_t)t * LL + bi) * PP + lane];
    if (lane == 0)
        act[s] = (bv > 0.0f) ? 1 : 0;
}

// ---------------- WMMA fragment loaders ----------------
// A (16x32 bf16): lanes 0-15 = row M, K {0..7,16..23}; lanes 16-31 same M rows, K {8..15,24..31}
__device__ __forceinline__ v16bf load_a_frag(const __bf16* base, int kc, int lane) {
    int m   = lane & 15;
    int off = (lane >> 4) * 8;
    const __bf16* p = base + m * XPAD + kc * 32 + off;
    union { v16bf v; uint4 q[2]; } u;
    u.q[0] = *(const uint4*)p;          // K off..off+7
    u.q[1] = *(const uint4*)(p + 16);   // K off+16..off+23
    return u.v;
}
// B (32x16 bf16): lanes 0-15 hold K {0..7,16..23} of column N=lane; lanes 16-31 hold K {8..15,24..31}.
// Weights stored column-major-per-permuted-column (k contiguous) -> two 16B chunks per lane (K off..off+7, off+16..off+23).
__device__ __forceinline__ v16bf load_b_frag(const __bf16* wm, int gt, int kc, int lane) {
    int n   = lane & 15;
    int off = (lane >> 4) * 8;
    const __bf16* p = wm + ((gt * 16 + n) * 128 + kc * 32 + off);
    union { v16bf v; uint4 q[2]; } u;
    u.q[0] = *(const uint4*)p;
    u.q[1] = *(const uint4*)(p + 16);
    return u.v;
}

// ---------------- kernel 2: fused gather + 9-step LSTM ----------------
__global__ __launch_bounds__(128) void lstm_kernel(
    const float* __restrict__ emb,
    const __bf16* __restrict__ wih, const __bf16* __restrict__ whh,
    const float* __restrict__ biasp, const int* __restrict__ nids,
    const int* __restrict__ act, float* __restrict__ out)
{
    __shared__ __bf16 xs[16 * XPAD];
    __shared__ __bf16 hs[16 * XPAD];

    const int tid  = threadIdx.x;
    const int lane = tid & 31;
    const int w    = tid >> 5;           // wave 0..3 -> hidden block of 32
    const int s0   = blockIdx.x * 16;

    for (int i = tid; i < 16 * XPAD; i += 128) hs[i] = (__bf16)0.0f;

    const int lane16 = lane & 15;
    const int gate   = lane & 3;         // 0:i 1:f 2:g 3:o
    const int u      = lane16 >> 2;      // hidden unit within tile
    const int rowme  = (lane >> 4) * 8;  // M of element 0 in D-fragment

    float biasr[8];
    #pragma unroll
    for (int nt = 0; nt < 8; ++nt)
        biasr[nt] = biasp[(w * 8 + nt) * 16 + lane16];

    v8f cst[8];
    #pragma unroll
    for (int nt = 0; nt < 8; ++nt) {
        v8f z = {0.f, 0.f, 0.f, 0.f, 0.f, 0.f, 0.f, 0.f};
        cst[nt] = z;
    }

    const int grow = tid >> 3;   // gather row 0..15
    const int gch  = tid & 7;    // gather chunk of 16 floats

    for (int p = 0; p < PP; ++p) {
        __syncthreads();                               // prior reads of xs done; hs ready
        {   // gather x_p -> LDS (fp32 -> bf16)
            int nid = nids[(s0 + grow) * PP + p];
            const float4* src = (const float4*)(emb + (size_t)nid * DD + gch * 16);
            float4 f0 = src[0], f1 = src[1], f2 = src[2], f3 = src[3];
            union { uint4 q[2]; __bf16 b[16]; } o;
            o.b[0]  = (__bf16)f0.x; o.b[1]  = (__bf16)f0.y; o.b[2]  = (__bf16)f0.z; o.b[3]  = (__bf16)f0.w;
            o.b[4]  = (__bf16)f1.x; o.b[5]  = (__bf16)f1.y; o.b[6]  = (__bf16)f1.z; o.b[7]  = (__bf16)f1.w;
            o.b[8]  = (__bf16)f2.x; o.b[9]  = (__bf16)f2.y; o.b[10] = (__bf16)f2.z; o.b[11] = (__bf16)f2.w;
            o.b[12] = (__bf16)f3.x; o.b[13] = (__bf16)f3.y; o.b[14] = (__bf16)f3.z; o.b[15] = (__bf16)f3.w;
            uint4* dst = (uint4*)&xs[grow * XPAD + gch * 16];
            dst[0] = o.q[0];
            dst[1] = o.q[1];
        }
        __syncthreads();                               // x ready

        // accumulators seeded with bias (C layout: N = lane -> per-lane scalar splat)
        v8f accv[8];
        #pragma unroll
        for (int nt = 0; nt < 8; ++nt) {
            float b = biasr[nt];
            v8f z = {b, b, b, b, b, b, b, b};
            accv[nt] = z;
        }
        #pragma unroll
        for (int kc = 0; kc < 4; ++kc) {
            v16bf ax = load_a_frag(xs, kc, lane);
            v16bf ah = load_a_frag(hs, kc, lane);
            #pragma unroll
            for (int nt = 0; nt < 8; ++nt) {
                int gt = w * 8 + nt;
                v16bf bA = load_b_frag(wih, gt, kc, lane);
                accv[nt] = __builtin_amdgcn_wmma_f32_16x16x32_bf16(
                    false, ax, false, bA, (short)0, accv[nt], false, false);
                v16bf bB = load_b_frag(whh, gt, kc, lane);
                accv[nt] = __builtin_amdgcn_wmma_f32_16x16x32_bf16(
                    false, ah, false, bB, (short)0, accv[nt], false, false);
            }
        }
        __syncthreads();                               // all GEMM reads of hs done

        const bool last = (p == PP - 1);
        #pragma unroll
        for (int nt = 0; nt < 8; ++nt) {
            const int j = w * 32 + nt * 4 + u;         // hidden index
            #pragma unroll
            for (int e = 0; e < 8; ++e) {
                float g = accv[nt][e];                 // bias already folded in
                float a = (gate == 2) ? fast_tanh(g) : fast_sigmoid(g);
                int base = lane & ~3;
                float vi = __shfl(a, base + 0, 32);
                float vf = __shfl(a, base + 1, 32);
                float vg = __shfl(a, base + 2, 32);
                float vo = __shfl(a, base + 3, 32);
                float cn = vf * cst[nt][e] + vi * vg;  // all quad lanes keep c in sync
                cst[nt][e] = cn;
                if (gate == 0) {                       // only role-0 lane computes/stores h
                    float h = vo * fast_tanh(cn);
                    int row = rowme + e;
                    hs[row * XPAD + j] = (__bf16)h;
                    if (last) {
                        int s = s0 + row;
                        out[(size_t)s * DD + j] = act[s] ? h : 0.0f;
                    }
                }
            }
        }
    }
}

// ---------------- launch ----------------
extern "C" void kernel_launch(void* const* d_in, const int* in_sizes, int n_in,
                              void* d_out, int out_size, void* d_ws, size_t ws_size,
                              hipStream_t stream)
{
    const float* cross = (const float*)d_in[0];
    const float* emb   = (const float*)d_in[1];
    const float* wih   = (const float*)d_in[2];
    const float* whh   = (const float*)d_in[3];
    const float* bih   = (const float*)d_in[4];
    const float* bhh   = (const float*)d_in[5];
    const int*   paths = (const int*)  d_in[6];
    float* out = (float*)d_out;
    char*  ws  = (char*)d_ws;

    __bf16* wbih  = (__bf16*)(ws + WS_WIH);
    __bf16* wbhh  = (__bf16*)(ws + WS_WHH);
    float*  obias = (float*) (ws + WS_BIAS);
    int*    nds   = (int*)   (ws + WS_NIDS);
    int*    actf  = (int*)   (ws + WS_ACT);

    prep_kernel<<<512, 256, 0, stream>>>(wih, whh, bih, bhh, wbih, wbhh, obias);
    leafpath_kernel<<<BT / 4, 128, 0, stream>>>(cross, paths, nds, actf);
    lstm_kernel<<<BT / 16, 128, 0, stream>>>(emb, wbih, wbhh, obias, nds, actf, out);
}
